// GroupedQueryAttention_38465727103117
// MI455X (gfx1250) — compile-verified, matrix operand order checked by reading, not run on hardware
//
#include <hip/hip_runtime.h>

// ---------------------------------------------------------------------------
// Fixed problem sizes (from the reference)
// ---------------------------------------------------------------------------
#define BB   2
#define SS   2048
#define HID  2048
#define NH   32
#define NKV  8
#define HD   64
#define GG   4          // NH / NKV
#define TOK  (BB * SS)  // 4096 tokens

typedef __bf16 bf16;
typedef __attribute__((ext_vector_type(4)))  __bf16 v4bf;
typedef __attribute__((ext_vector_type(8)))  __bf16 v8bf;
typedef __attribute__((ext_vector_type(16))) __bf16 v16bf;
typedef __attribute__((ext_vector_type(4)))  float  v4f;
typedef __attribute__((ext_vector_type(8)))  float  v8f;

// ---------------------------------------------------------------------------
// Fragment loader: A/B operand of v_wmma_f32_16x16x32_bf16.
// 16 rows x 32 K, row-major source with leading dim `ld` (elements).
// Lanes 0-15 row=lane: halves 0..7 = K0..7, 8..15 = K16..23;
// lanes 16-31 row=lane-16: K8..15 / K24..31.  Two b128 loads per lane.
// ---------------------------------------------------------------------------
__device__ __forceinline__ v16bf load_frag(const bf16* __restrict__ p, int ld) {
  const int lane = threadIdx.x & 31;
  const int r  = lane & 15;
  const int hh = lane >> 4;
  const bf16* base = p + (size_t)r * ld + hh * 8;
  v8bf lo = *(const v8bf*)(base);
  v8bf hi = *(const v8bf*)(base + 16);
  v16bf f;
#pragma unroll
  for (int i = 0; i < 8; ++i) { f[i] = lo[i]; f[8 + i] = hi[i]; }
  return f;
}

__device__ __forceinline__ v8f wmma_bf16(v16bf a, v16bf b, v8f c) {
  return __builtin_amdgcn_wmma_f32_16x16x32_bf16(
      false, a, false, b, (short)0, c, false, false);
}

// ---------------------------------------------------------------------------
// VALU butterfly reductions within 16-lane rows via v_permlane16_b32.
// Lane-select constants encode sel(i) = i ^ mask (nibble per lane).
// ---------------------------------------------------------------------------
__device__ __forceinline__ float pl16(float v, unsigned lo, unsigned hi) {
  unsigned u = __float_as_uint(v);
  unsigned r = __builtin_amdgcn_permlane16(u, u, lo, hi, false, false);
  return __uint_as_float(r);
}
__device__ __forceinline__ float redmax16(float v) {
  v = fmaxf(v, pl16(v, 0x67452301u, 0xEFCDAB89u));  // xor 1
  v = fmaxf(v, pl16(v, 0x54761032u, 0xDCFE98BAu));  // xor 2
  v = fmaxf(v, pl16(v, 0x32107654u, 0xBA98FEDCu));  // xor 4
  v = fmaxf(v, pl16(v, 0xFEDCBA98u, 0x76543210u));  // xor 8
  return v;
}
__device__ __forceinline__ float redsum16(float v) {
  v += pl16(v, 0x67452301u, 0xEFCDAB89u);
  v += pl16(v, 0x54761032u, 0xDCFE98BAu);
  v += pl16(v, 0x32107654u, 0xBA98FEDCu);
  v += pl16(v, 0xFEDCBA98u, 0x76543210u);
  return v;
}

// ---------------------------------------------------------------------------
// fp32 -> bf16 conversion, 4 elements per thread (b128 in, b64 out)
// ---------------------------------------------------------------------------
__global__ void f32_to_bf16_kernel(const float* __restrict__ in,
                                   bf16* __restrict__ out, int n4) {
  int i = blockIdx.x * blockDim.x + threadIdx.x;
  int stride = gridDim.x * blockDim.x;
  for (; i < n4; i += stride) {
    v4f v = *(const v4f*)(in + (size_t)i * 4);
    v4bf o;
#pragma unroll
    for (int j = 0; j < 4; ++j) o[j] = (bf16)v[j];
    *(v4bf*)(out + (size_t)i * 4) = o;
  }
}

// ---------------------------------------------------------------------------
// NT GEMM: C[M,N] = A[M,K] @ B[N,K]^T.  One wave: 32(M) x 64(N) tile,
// 8 accumulators, 8 WMMAs per K-step of 32 (2 A-frags + 4 B-frags loads).
// MODE 0: bf16 [M,N]
// MODE 1: fp32 [M,N]
// MODE 2: bf16 V^T [B,NKV,HD,S]                          (N == 512)
// MODE 3: bf16 [M,N] + fused RoPE + post-scale.  The 64-wide N-tile is
//         exactly one head (HD=64): rotation pair (d, d+32) = accumulator
//         tiles (t, t+2) at the same r, so RoPE runs on fp32 accums in regs.
// ---------------------------------------------------------------------------
template <int MODE>
__global__ void gemm_nt_kernel(const bf16* __restrict__ A,
                               const bf16* __restrict__ Bm,
                               void* __restrict__ outp,
                               int M, int N, int K, float postscale) {
  const int lane = threadIdx.x & 31;
  const int wv   = threadIdx.x >> 5;
  const int tilesN = N >> 6;
  const int tilesM = M >> 5;
  const int job = blockIdx.x * 8 + wv;
  if (job >= tilesM * tilesN) return;
  const int mT = job / tilesN;
  const int nT = job % tilesN;
  const int col = lane & 15;
  const int hh  = lane >> 4;

  v8f acc[2][4] = {};
  const bf16* aP0 = A + (size_t)mT * 32 * K;
  const bf16* aP1 = aP0 + (size_t)16 * K;
  const bf16* bP  = Bm + (size_t)nT * 64 * K;

#pragma unroll 2
  for (int kb = 0; kb < K; kb += 32) {
    __builtin_prefetch(aP0 + kb + 512, 0, 1);
    v16bf a0 = load_frag(aP0 + kb, K);
    v16bf a1 = load_frag(aP1 + kb, K);
#pragma unroll
    for (int t = 0; t < 4; ++t) {
      v16bf b = load_frag(bP + (size_t)t * 16 * K + kb, K);
      acc[0][t] = wmma_bf16(a0, b, acc[0][t]);
      acc[1][t] = wmma_bf16(a1, b, acc[1][t]);
    }
  }

  if constexpr (MODE == 3) {
    // fused RoPE epilogue: head = nT, d = t*16+col (t=0,1), pair at t+2
    bf16* out = (bf16*)outp;
#pragma unroll
    for (int t = 0; t < 2; ++t) {
      const int d = t * 16 + col;
      const float invf = __powf(10000.0f, -(float)d * (1.0f / 32.0f));
#pragma unroll
      for (int sub = 0; sub < 2; ++sub) {
#pragma unroll
        for (int r = 0; r < 8; ++r) {
          const int m = mT * 32 + sub * 16 + r + 8 * hh;
          const int s = m & (SS - 1);
          const float ang = (float)s * invf;
          const float c  = __cosf(ang);
          const float sn = __sinf(ang);
          const float x1 = acc[sub][t][r];
          const float x2 = acc[sub][t + 2][r];
          bf16* row = out + (size_t)m * N + nT * 64 + d;
          row[0]  = (bf16)((x1 * c - x2 * sn) * postscale);
          row[32] = (bf16)((x2 * c + x1 * sn) * postscale);
        }
      }
    }
  } else {
#pragma unroll
    for (int sub = 0; sub < 2; ++sub) {
      if constexpr (MODE == 0) {
        bf16* out = (bf16*)outp;
#pragma unroll
        for (int t = 0; t < 4; ++t) {
          const int n = nT * 64 + t * 16 + col;
#pragma unroll
          for (int r = 0; r < 8; ++r)
            out[(size_t)(mT * 32 + sub * 16 + r + 8 * hh) * N + n] =
                (bf16)acc[sub][t][r];
        }
      } else if constexpr (MODE == 1) {
        float* out = (float*)outp;
#pragma unroll
        for (int t = 0; t < 4; ++t) {
          const int n = nT * 64 + t * 16 + col;
#pragma unroll
          for (int r = 0; r < 8; ++r)
            out[(size_t)(mT * 32 + sub * 16 + r + 8 * hh) * N + n] =
                (float)acc[sub][t][r];
        }
      } else {  // MODE == 2: V^T store, N = NKV*HD = 512
        bf16* vt = (bf16*)outp;
#pragma unroll
        for (int t = 0; t < 4; ++t) {
          const int n   = nT * 64 + t * 16 + col;
          const int kvh = n >> 6;
          const int d   = n & 63;
#pragma unroll
          for (int r = 0; r < 8; ++r) {
            const int m = mT * 32 + sub * 16 + r + 8 * hh;
            const int b = m >> 11;
            const int s = m & (SS - 1);
            vt[((size_t)(b * NKV + kvh) * HD + d) * SS + s] =
                (bf16)acc[sub][t][r];
          }
        }
      }
    }
  }
}

// ---------------------------------------------------------------------------
// Flash attention. One wave per 16-row query tile per (b, h). Key blocks of
// 64: 8 WMMAs for scores, online softmax (permlane16 reductions, exp), P tile
// routed C-layout -> A-layout through per-wave LDS, 8 WMMAs for P·V.
// Q: bf16 [B,S,NH,HD] (pre-scaled by 1/8); K: bf16 [B,S,NKV,HD];
// Vt: bf16 [B,NKV,HD,S]; ctx: bf16 [B,S,NH,HD].
// ---------------------------------------------------------------------------
__global__ void attn_kernel(const bf16* __restrict__ q,
                            const bf16* __restrict__ k,
                            const bf16* __restrict__ vt,
                            bf16* __restrict__ ctx) {
  __shared__ __attribute__((aligned(16))) bf16 sP[8][16][64];

  const int lane = threadIdx.x & 31;
  const int wv   = threadIdx.x >> 5;
  const int job  = blockIdx.x * 8 + wv;       // [0, B*NH*(S/16))
  const int qt   = job & 127;                 // S/16 = 128
  const int h    = (job >> 7) & (NH - 1);
  const int b    = job >> 12;
  const int kvh  = h >> 2;
  const int col  = lane & 15;
  const int hh   = lane >> 4;

  const bf16* qbase = q + ((size_t)(b * SS + qt * 16) * NH + h) * HD;
  const v16bf a0 = load_frag(qbase, NH * HD);
  const v16bf a1 = load_frag(qbase + 32, NH * HD);

  v8f o[4] = {};
  float mrow[8], lrow[8];
#pragma unroll
  for (int r = 0; r < 8; ++r) { mrow[r] = -__builtin_inff(); lrow[r] = 0.0f; }

  const bf16* vhead = vt + ((size_t)(b * NKV + kvh) * HD) * SS;
  bf16* const pbase = &sP[wv][0][0];
  const int kld = NKV * HD;                   // 512

  for (int kb = 0; kb < SS; kb += 64) {
    // ---- scores 16x64: 8 WMMAs ----
    const bf16* kbase = k + ((size_t)(b * SS + kb) * NKV + kvh) * HD;
    v8f s[4];
#pragma unroll
    for (int t = 0; t < 4; ++t) {
      v16bf bk0 = load_frag(kbase + (size_t)t * 16 * kld, kld);
      v16bf bk1 = load_frag(kbase + (size_t)t * 16 * kld + 32, kld);
      v8f z = {};
      z = wmma_bf16(a0, bk0, z);
      s[t] = wmma_bf16(a1, bk1, z);
    }

    // ---- online softmax (Q pre-scaled; rows = (r, half) per lane) ----
    float lm[8];
#pragma unroll
    for (int r = 0; r < 8; ++r)
      lm[r] = redmax16(fmaxf(fmaxf(s[0][r], s[1][r]), fmaxf(s[2][r], s[3][r])));

    float corr[8], ls[8];
#pragma unroll
    for (int r = 0; r < 8; ++r) {
      const float mn = fmaxf(mrow[r], lm[r]);
      corr[r] = __expf(mrow[r] - mn);
      mrow[r] = mn;
      ls[r] = 0.0f;
    }
#pragma unroll
    for (int t = 0; t < 4; ++t) {
#pragma unroll
      for (int r = 0; r < 8; ++r) {
        const float p = __expf(s[t][r] - mrow[r]);
        ls[r] += p;
        sP[wv][r + 8 * hh][t * 16 + col] = (bf16)p;
      }
    }
#pragma unroll
    for (int r = 0; r < 8; ++r) {
      lrow[r] = lrow[r] * corr[r] + redsum16(ls[r]);
#pragma unroll
      for (int t = 0; t < 4; ++t) o[t][r] *= corr[r];
    }

    // ---- P: C-layout (LDS) -> two A-operand fragments (K 0..31 / 32..63) ----
    v16bf pf0 = load_frag(pbase, 64);
    v16bf pf1 = load_frag(pbase + 32, 64);

    // ---- P(16x64) @ V(64x64) via Vt (K-major contiguous): 8 WMMAs ----
    const bf16* vbase = vhead + kb;
#pragma unroll
    for (int t = 0; t < 4; ++t) {
      v16bf v0 = load_frag(vbase + (size_t)t * 16 * SS, SS);
      v16bf v1 = load_frag(vbase + (size_t)t * 16 * SS + 32, SS);
      o[t] = wmma_bf16(pf0, v0, o[t]);
      o[t] = wmma_bf16(pf1, v1, o[t]);
    }
  }

  // ---- epilogue: O /= l, store ctx bf16 [B,S,NH,HD] ----
  bf16* cbase = ctx + ((size_t)(b * SS + qt * 16) * NH + h) * HD;
#pragma unroll
  for (int r = 0; r < 8; ++r) {
    const float inv = 1.0f / lrow[r];
    const size_t rowoff = (size_t)(r + 8 * hh) * (NH * HD);
#pragma unroll
    for (int t = 0; t < 4; ++t)
      cbase[rowoff + t * 16 + col] = (bf16)(o[t][r] * inv);
  }
}

// ---------------------------------------------------------------------------
// Host-side orchestration
// ---------------------------------------------------------------------------
extern "C" void kernel_launch(void* const* d_in, const int* in_sizes, int n_in,
                              void* d_out, int out_size, void* d_ws, size_t ws_size,
                              hipStream_t stream) {
  const float* hs = (const float*)d_in[0];
  const float* wq = (const float*)d_in[1];
  const float* wk = (const float*)d_in[2];
  const float* wv = (const float*)d_in[3];
  const float* wo = (const float*)d_in[4];
  float* out = (float*)d_out;

  char* ws = (char*)d_ws;
  bf16* hsb = (bf16*)(ws + 0);          // 16 MB  [TOK, HID]
  bf16* wqb = (bf16*)(ws + 16777216);   //  8 MB  [HID, HID]
  bf16* wkb = (bf16*)(ws + 25165824);   //  2 MB  [512, HID]
  bf16* wvb = (bf16*)(ws + 27262976);   //  2 MB  [512, HID]
  bf16* wob = (bf16*)(ws + 29360128);   //  8 MB  [HID, HID]
  bf16* qb  = (bf16*)(ws + 37748736);   // 16 MB  [B,S,NH,HD]  (RoPE'd, /8)
  bf16* kbf = (bf16*)(ws + 54525952);   //  4 MB  [B,S,NKV,HD] (RoPE'd)
  bf16* vtb = (bf16*)(ws + 58720256);   //  4 MB  [B,NKV,HD,S]
  bf16* ctx = (bf16*)(ws + 62914560);   // 16 MB  [B,S,NH,HD]

  // 1) fp32 -> bf16 (vectorized x4)
  f32_to_bf16_kernel<<<2048, 256, 0, stream>>>(hs, hsb, TOK * HID / 4);
  f32_to_bf16_kernel<<<1024, 256, 0, stream>>>(wq, wqb, HID * HID / 4);
  f32_to_bf16_kernel<<<256,  256, 0, stream>>>(wk, wkb, NKV * HD * HID / 4);
  f32_to_bf16_kernel<<<256,  256, 0, stream>>>(wv, wvb, NKV * HD * HID / 4);
  f32_to_bf16_kernel<<<1024, 256, 0, stream>>>(wo, wob, HID * HID / 4);

  // 2) projections (WMMA bf16 GEMMs); RoPE fused into Q/K epilogues,
  //    1/sqrt(HD) folded into Q, V stored transposed for the PV WMMA.
  gemm_nt_kernel<3><<<512, 256, 0, stream>>>(hsb, wqb, qb,  TOK, HID, HID, 0.125f);
  gemm_nt_kernel<3><<<128, 256, 0, stream>>>(hsb, wkb, kbf, TOK, NKV * HD, HID, 1.0f);
  gemm_nt_kernel<2><<<128, 256, 0, stream>>>(hsb, wvb, vtb, TOK, NKV * HD, HID, 1.0f);

  // 3) flash attention (WMMA scores + WMMA PV, permlane16 softmax)
  attn_kernel<<<1024, 256, 0, stream>>>(qb, kbf, vtb, ctx);

  // 4) output projection, fp32 straight to d_out
  gemm_nt_kernel<1><<<512, 256, 0, stream>>>(ctx, wob, (void*)out, TOK, HID, HID, 1.0f);
}
